// VQLayerSG_9947144257864
// MI455X (gfx1250) — compile-verified, matrix operand order checked by reading, not run on hardware
//
#include <hip/hip_runtime.h>
#include <hip/hip_bf16.h>
#include <stdint.h>

#define Bsz   32
#define Fdim  256
#define Tdim  2048
#define Kcb   1024
#define NKT   64            // K / 16 k-tiles
#define TILE_BYTES 8192     // one k-tile (8 chunks * 32 lanes * 32B), per hi or lo
#define LDS_BUF (2*TILE_BYTES)
#define NOUT  (Bsz*Fdim*Tdim)

typedef __attribute__((ext_vector_type(16))) __bf16 v16bf;
typedef __attribute__((ext_vector_type(8)))  float  v8f;

// CDNA5 async global->LDS copy (ASYNCcnt), 16 bytes per lane.
__device__ __forceinline__ void async_b128(uint32_t lds_off, const void* g) {
  asm volatile("global_load_async_to_lds_b128 %0, %1, off"
               :: "v"(lds_off), "v"(g) : "memory");
}

// f-index inside a 32-wide chunk for WMMA 16-bit A/B per-lane layout
__device__ __forceinline__ int fmap(int j, int h) {
  return (j < 8) ? (h*8 + j) : (16 + h*8 + (j - 8));
}

// ---------------- repack codebook into WMMA-layout bf16 hi/lo tiles ----------
__global__ __launch_bounds__(256) void vq_repack_kernel(
    const float* __restrict__ w, __bf16* __restrict__ bhi, __bf16* __restrict__ blo)
{
  const int gid = blockIdx.x * 256 + threadIdx.x;   // 0..16383 = 64 kt * 8 c * 32 lanes
  const int laneInTile = gid & 31;
  const int c  = (gid >> 5) & 7;
  const int kt = gid >> 8;
  const int n  = laneInTile & 15;
  const int h  = laneInTile >> 4;
  const float* row = w + (size_t)(kt*16 + n) * Fdim;
  const size_t base = ((size_t)(kt*8 + c)) * 512 + (size_t)laneInTile * 16;
#pragma unroll
  for (int j = 0; j < 16; ++j) {
    const int f = c*32 + fmap(j, h);
    float v = row[f];
    __bf16 hi = (__bf16)v;
    __bf16 lo = (__bf16)(v - (float)hi);
    bhi[base + j] = hi;
    blo[base + j] = lo;
  }
}

// ---------------- |c|^2 per code ---------------------------------------------
__global__ __launch_bounds__(256) void vq_c2_kernel(
    const float* __restrict__ w, float* __restrict__ c2)
{
  const int k = blockIdx.x * 256 + threadIdx.x;     // 0..1023
  const float* row = w + (size_t)k * Fdim;
  float s = 0.f;
  for (int f = 0; f < Fdim; ++f) s = fmaf(row[f], row[f], s);
  c2[k] = s;
}

// ---------------- main: distances + argmin via WMMA --------------------------
// Each wave owns TWO 16-row t-tiles (M=32): two independent accumulator chains
// share every B tile fetched from LDS -> 2x WMMA per DS byte, 2x XDL ILP.
__global__ __launch_bounds__(256) void vq_main_kernel(
    const float* __restrict__ x, const __bf16* __restrict__ bhi,
    const __bf16* __restrict__ blo, const float* __restrict__ c2,
    float* __restrict__ mind, int* __restrict__ midx)
{
  __shared__ alignas(32) char lds[2 * LDS_BUF];     // 32 KB: double-buffered hi+lo tile
  const int tid  = threadIdx.x;
  const int lane = tid & 31;
  const int m    = lane & 15;                       // N (code in tile) / M row index
  const int h    = lane >> 4;                       // lane half
  const int wave = tid >> 5;
  const int tile = blockIdx.x * 8 + wave;           // 0..2047 (32-row t-tiles)
  const int b    = tile >> 6;                       // / (T/32)
  const int t0   = (tile & 63) << 5;                // 32 rows per wave

  // ---- load A tiles (2 x 16 t x 256 f), bf16 hi/lo split, |a|^2 on the fly --
  v16bf ahi[2][8], alo[2][8];
  float a2v[2][8];
#pragma unroll
  for (int s = 0; s < 2; ++s) {
    const float* xb = x + (size_t)b * ((size_t)Fdim * Tdim) + (t0 + s*16 + m);
    float a2p = 0.f;
#pragma unroll
    for (int c = 0; c < 8; ++c) {
#pragma unroll
      for (int j = 0; j < 16; ++j) {
        const int f = c*32 + fmap(j, h);
        float v = xb[(size_t)f * Tdim];
        __bf16 hi = (__bf16)v;
        __bf16 lo = (__bf16)(v - (float)hi);
        ahi[s][c][j] = hi;
        alo[s][c][j] = lo;
        a2p = fmaf(v, v, a2p);
      }
    }
    // lane L and L^16 hold complementary f subsets of row m -> combine halves
    float a2row = a2p + __shfl_xor(a2p, 16, 32);
#pragma unroll
    for (int i = 0; i < 8; ++i) a2v[s][i] = __shfl(a2row, h*8 + i, 32); // row M=h*8+i
  }

  const uint32_t ldsbase = (uint32_t)(uintptr_t)&lds[0];
  const char* gh = (const char*)bhi;
  const char* gl = (const char*)blo;

  // prologue: async-stage k-tile 0 into buffer 0 (4 x b128 per thread)
  {
    uint32_t d0 = ldsbase + tid*32;
    const char* s0 = gh + tid*32;
    async_b128(d0,      s0);
    async_b128(d0 + 16, s0 + 16);
    uint32_t d1 = d0 + TILE_BYTES;
    const char* s1 = gl + tid*32;
    async_b128(d1,      s1);
    async_b128(d1 + 16, s1 + 16);
  }

  float minv[2][8]; int mini[2][8];
#pragma unroll
  for (int s = 0; s < 2; ++s)
#pragma unroll
    for (int i = 0; i < 8; ++i) { minv[s][i] = 3.4e38f; mini[s][i] = 0; }

  for (int kt = 0; kt < NKT; ++kt) {
    const int cur = kt & 1;
    if (kt + 1 < NKT) {                              // prefetch next tile
      uint32_t d0 = ldsbase + (cur ^ 1)*LDS_BUF + tid*32;
      const char* s0 = gh + (size_t)(kt+1)*TILE_BYTES + tid*32;
      async_b128(d0,      s0);
      async_b128(d0 + 16, s0 + 16);
      uint32_t d1 = d0 + TILE_BYTES;
      const char* s1 = gl + (size_t)(kt+1)*TILE_BYTES + tid*32;
      async_b128(d1, s1);
      async_b128(d1 + 16, s1 + 16);
      asm volatile("s_wait_asynccnt 4" ::: "memory"); // older 4 (current tile) done
    } else {
      asm volatile("s_wait_asynccnt 0" ::: "memory");
    }
    __syncthreads();

    const char* basep = &lds[cur * LDS_BUF];
    v8f acc0 = {}, acc1 = {};
#pragma unroll
    for (int c = 0; c < 8; ++c) {
      v16bf bh = *(const v16bf*)(basep + c*1024 + lane*32);
      v16bf bl = *(const v16bf*)(basep + TILE_BYTES + c*1024 + lane*32);
      // hi*hi + hi*lo + lo*hi : ~fp32-accurate cross term, f32 accumulate.
      // Two independent chains (acc0/acc1) interleave on the matrix pipe.
      acc0 = __builtin_amdgcn_wmma_f32_16x16x32_bf16(false, ahi[0][c], false, bh, (short)0, acc0, false, false);
      acc1 = __builtin_amdgcn_wmma_f32_16x16x32_bf16(false, ahi[1][c], false, bh, (short)0, acc1, false, false);
      acc0 = __builtin_amdgcn_wmma_f32_16x16x32_bf16(false, ahi[0][c], false, bl, (short)0, acc0, false, false);
      acc1 = __builtin_amdgcn_wmma_f32_16x16x32_bf16(false, ahi[1][c], false, bl, (short)0, acc1, false, false);
      acc0 = __builtin_amdgcn_wmma_f32_16x16x32_bf16(false, alo[0][c], false, bh, (short)0, acc0, false, false);
      acc1 = __builtin_amdgcn_wmma_f32_16x16x32_bf16(false, alo[1][c], false, bh, (short)0, acc1, false, false);
    }
    const float c2v  = c2[kt*16 + m];
    const int   kidx = kt*16 + m;
#pragma unroll
    for (int i = 0; i < 8; ++i) {
      float d0 = (a2v[0][i] - 2.0f*acc0[i] + c2v) * (1.0f / (float)Fdim);
      if (d0 < minv[0][i]) { minv[0][i] = d0; mini[0][i] = kidx; }
      float d1 = (a2v[1][i] - 2.0f*acc1[i] + c2v) * (1.0f / (float)Fdim);
      if (d1 < minv[1][i]) { minv[1][i] = d1; mini[1][i] = kidx; }
    }
    __syncthreads();                                 // all waves done reading cur
  }

  // min/argmin across the 16 lanes of each half (k dimension), first-index ties
#pragma unroll
  for (int s = 0; s < 2; ++s) {
#pragma unroll
    for (int i = 0; i < 8; ++i) {
      float v = minv[s][i]; int ix = mini[s][i];
#pragma unroll
      for (int st = 8; st >= 1; st >>= 1) {
        float ov = __shfl_xor(v,  st, 32);
        int   oi = __shfl_xor(ix, st, 32);
        if (ov < v || (ov == v && oi < ix)) { v = ov; ix = oi; }
      }
      minv[s][i] = v; mini[s][i] = ix;
    }
  }
  if (m == 0) {                                      // lanes 0 and 16 write 8 rows each
#pragma unroll
    for (int s = 0; s < 2; ++s)
#pragma unroll
      for (int i = 0; i < 8; ++i) {
        const int t = t0 + s*16 + h*8 + i;
        mind[b*Tdim + t] = minv[s][i];
        midx[b*Tdim + t] = mini[s][i];
      }
  }
}

// ---------------- deterministic loss reduction -------------------------------
__global__ __launch_bounds__(256) void vq_loss_kernel(
    const float* __restrict__ mind, float* __restrict__ out)
{
  __shared__ float red[256];
  const int tid = threadIdx.x;
  float s = 0.f;
  const int per = (Bsz*Tdim) / 256;                  // 256
  for (int j = 0; j < per; ++j) s += mind[tid*per + j];
  red[tid] = s;
  __syncthreads();
  for (int step = 128; step > 0; step >>= 1) {
    if (tid < step) red[tid] += red[tid + step];
    __syncthreads();
  }
  if (tid == 0) {
    float loss = red[0] / (float)(Bsz*Tdim);
    out[NOUT]     = loss;   // loss_commit
    out[NOUT + 1] = loss;   // loss_codebook (numerically identical forward)
  }
}

// ---------------- gather out[b,:,t] = w[idx[b,t],:] --------------------------
__global__ __launch_bounds__(256) void vq_gather_kernel(
    const float* __restrict__ w, const int* __restrict__ midx, float* __restrict__ out)
{
  const int blk = blockIdx.x;                        // 0..255
  const int b = blk >> 3;
  const int t = ((blk & 7) << 8) + threadIdx.x;
  const int k = midx[b*Tdim + t];
  const float4* wr = (const float4*)(w + (size_t)k * Fdim);
  float* ob = out + (size_t)b * ((size_t)Fdim * Tdim) + t;
  for (int f4 = 0; f4 < Fdim/4; ++f4) {
    float4 v = wr[f4];
    ob[(size_t)(4*f4 + 0) * Tdim] = v.x;             // coalesced across lanes (t)
    ob[(size_t)(4*f4 + 1) * Tdim] = v.y;
    ob[(size_t)(4*f4 + 2) * Tdim] = v.z;
    ob[(size_t)(4*f4 + 3) * Tdim] = v.w;
  }
}

extern "C" void kernel_launch(void* const* d_in, const int* in_sizes, int n_in,
                              void* d_out, int out_size, void* d_ws, size_t ws_size,
                              hipStream_t stream) {
  const float* x = (const float*)d_in[0];            // [B,F,T] fp32
  const float* w = (const float*)d_in[1];            // [K,F]   fp32
  float* out = (float*)d_out;                        // [B,F,T] + 2 scalars

  char* ws = (char*)d_ws;
  __bf16* bhi = (__bf16*)(ws);                       // 512 KB
  __bf16* blo = (__bf16*)(ws + 524288);              // 512 KB
  float*  c2  = (float*) (ws + 1048576);             // 4 KB
  float*  mind= (float*) (ws + 1048576 + 4096);      // 256 KB
  int*    midx= (int*)   (ws + 1048576 + 4096 + 262144); // 256 KB

  vq_repack_kernel<<<64, 256, 0, stream>>>(w, bhi, blo);
  vq_c2_kernel<<<4, 256, 0, stream>>>(w, c2);
  vq_main_kernel<<<256, 256, 0, stream>>>(x, bhi, blo, c2, mind, midx);
  vq_loss_kernel<<<1, 256, 0, stream>>>(mind, out);
  vq_gather_kernel<<<256, 256, 0, stream>>>(w, midx, out);
}
